// LSTMModel_14654428414241
// MI455X (gfx1250) — matrix operand order repaired and checked by reading, not compiled, not run on hardware
//
#include <hip/hip_runtime.h>
#include <stdint.h>

// Problem constants (match reference)
#define H      128
#define IN0    93
#define BATCH  256
#define TSEQ   512
// Tiling
#define BM      16              // batch rows per workgroup (WMMA M)
#define NWAVES  8               // waves per workgroup: 8 * 16 = 128 hidden
#define THREADS (NWAVES * 32)

typedef __attribute__((ext_vector_type(16))) __bf16 v16bf;
typedef __attribute__((ext_vector_type(8)))  float  v8f;

struct B32 { uint4 lo, hi; };   // 32 bytes == one v16bf

__device__ __forceinline__ uint16_t f2bf_u16(float f) {
  uint32_t u = __builtin_bit_cast(uint32_t, f);
  u = (u + 0x7FFFu + ((u >> 16) & 1u)) >> 16;   // round-to-nearest-even
  return (uint16_t)u;
}
__device__ __forceinline__ float bf2f(uint16_t s) {
  uint32_t u = ((uint32_t)s) << 16;
  return __builtin_bit_cast(float, u);
}
__device__ __forceinline__ float sigmf(float x) {
  return 1.0f / (1.0f + __expf(-x));
}

// ---------------------------------------------------------------------------
// Prep: x fp32 [B,T,IN0] (batch-major) -> bf16 [T,B,H] time-major, K zero-
// padded 93 -> 128. Runs once; ~80 MB of traffic, trivial at 23.3 TB/s.
// Keeps all fp32->bf16 conversion of x OFF the sequential recurrence path.
// ---------------------------------------------------------------------------
__global__ __launch_bounds__(256)
void x_to_bf16(const float* __restrict__ x, uint16_t* __restrict__ xb)
{
  const size_t idx = (size_t)blockIdx.x * blockDim.x + threadIdx.x;
  const size_t total = (size_t)TSEQ * BATCH * H;
  if (idx >= total) return;
  const int k = (int)(idx & (H - 1));
  const size_t tb = idx >> 7;              // t*BATCH + b
  const int b = (int)(tb & (BATCH - 1));
  const int t = (int)(tb >> 8);
  uint16_t v = 0;
  if (k < IN0) v = f2bf_u16(x[((size_t)b * TSEQ + t) * IN0 + k]);
  xb[idx] = v;
}

// ---------------------------------------------------------------------------
// One LSTM layer. grid.x = BATCH/BM workgroups; each owns a 16-row batch tile
// for the entire T-step recurrence (batch is the only recurrence-parallel dim).
// Input is always bf16 [T,B,H] time-major. Per wave, per step:
//   16 WMMAs (x_t @ W_ih^T, B streamed from LDS) +
//   16 WMMAs (h_{t-1} @ W_hh^T, B resident in 128 VGPRs),
// then a pure-VALU i/f/g/o update (all four gate tiles share lane layout).
// The x tile for step t+1 is software-pipelined: its global loads are issued
// before step t's WMMA stream so L2 latency is hidden behind the matrix ops.
// h is double-buffered in LDS -> exactly one barrier per timestep.
// ---------------------------------------------------------------------------
__global__ __launch_bounds__(THREADS)
void lstm_layer(const uint16_t* __restrict__ xin,           // bf16 [T,B,H]
                const float* __restrict__ Wih, int Kin,     // [4H, Kin]
                const float* __restrict__ Whh,              // [4H, H]
                const float* __restrict__ bih,              // [4H]
                const float* __restrict__ bhh,              // [4H]
                uint16_t* __restrict__ hout)                // bf16 [T,B,H]
{
  extern __shared__ char smem[];
  // W_ih^T staged in WMMA B-chunk layout: 32 N-tiles * 4 K-chunks, 1KB each.
  uint16_t* sWih = (uint16_t*)smem;                    // 65536 u16 = 128 KB
  uint16_t* sH   = (uint16_t*)(smem + 131072);         // 2 * BM * H u16 = 8 KB

  const int tid   = threadIdx.x;
  const int lane  = tid & 31;
  const int wave  = tid >> 5;          // 0..7 -> hidden slice
  const int m0    = blockIdx.x * BM;   // batch tile base
  const int n0    = wave * 16;         // hidden slice base
  const int col   = lane & 15;         // C/D tile: N column of this lane
  const int khalf = lane >> 4;         // A layout: K-half selector

  // ---- Prologue: swizzle W_ih (fp32, [4H][Kin]) into LDS as bf16 B-chunks.
  // Chunk c = ntile*4 + kc holds B[k][n] at element lane*16 + s (n = lane&15,
  // k in WMMA table order for half lane>>4), so each
  // lane's B operand is one contiguous 32-byte run (2x ds_load_b128).
  for (int i = tid; i < 128 * 512; i += THREADS) {
    const int c  = i >> 9;
    const int e  = i & 511;
    const int kl = ((e & 8) << 1) + ((e >> 8) << 3) + (e & 7);  // k within chunk
    const int nn = (e >> 4) & 15;      // n within tile
    const int ntile = c >> 2, kc = c & 3;
    const int n = ntile * 16 + nn;     // absolute gate-output index (0..511)
    const int k = kc * 32 + kl;
    const float w = (k < Kin) ? Wih[n * Kin + k] : 0.0f;  // zero-pad K (layer 0)
    sWih[c * 512 + e] = f2bf_u16(w);
  }
  // h_{-1} = 0
  for (int i = tid; i < 2 * BM * H; i += THREADS) sH[i] = 0;

  // ---- W_hh^T B-chunks resident in VGPRs for the whole recurrence.
  v16bf Bhh[4][4];
#pragma unroll
  for (int q = 0; q < 4; ++q)
#pragma unroll
    for (int kc = 0; kc < 4; ++kc) {
      v16bf b;
#pragma unroll
      for (int j = 0; j < 16; ++j) {
        const int N = q * 128 + n0 + col;
        const int kk = kc * 32 + ((j & 8) << 1) + khalf * 8 + (j & 7);
        b[j] = __builtin_bit_cast(__bf16, f2bf_u16(Whh[N * H + kk]));
      }
      Bhh[q][kc] = b;
    }

  float bias[4];
#pragma unroll
  for (int q = 0; q < 4; ++q)
    bias[q] = bih[q * 128 + n0 + col] + bhh[q * 128 + n0 + col];

  v8f cst = {0.f, 0.f, 0.f, 0.f, 0.f, 0.f, 0.f, 0.f};   // cell state in registers

  __syncthreads();

  const int mrow = m0 + col;   // A-matrix: this lane's batch row (M = lane&15)

  // ---- Software-pipelined x tile: preload step 0.
  B32 axraw[4];
  {
    const uint16_t* xr = xin + ((size_t)0 * BATCH + mrow) * H;
#pragma unroll
    for (int kc = 0; kc < 4; ++kc) {
      axraw[kc].lo = *(const uint4*)(xr + kc * 32 + khalf * 8);
      axraw[kc].hi = *(const uint4*)(xr + kc * 32 + 16 + khalf * 8);
    }
  }

  for (int t = 0; t < TSEQ; ++t) {
    // consume the prefetched x tile
    v16bf Ax[4];
#pragma unroll
    for (int kc = 0; kc < 4; ++kc) Ax[kc] = __builtin_bit_cast(v16bf, axraw[kc]);

    // issue next step's x loads now; they drain behind this step's WMMAs
    {
      const int tn = (t + 1 < TSEQ) ? (t + 1) : t;   // branch-free clamp
      const uint16_t* xr = xin + ((size_t)tn * BATCH + mrow) * H;
#pragma unroll
      for (int kc = 0; kc < 4; ++kc) {
        axraw[kc].lo = *(const uint4*)(xr + kc * 32 + khalf * 8);
        axraw[kc].hi = *(const uint4*)(xr + kc * 32 + 16 + khalf * 8);
      }
    }

    // ---- gates = x_t @ W_ih^T  (16 WMMAs; B operands streamed from LDS)
    v8f acc[4];
#pragma unroll
    for (int q = 0; q < 4; ++q) {
      v8f a = {0.f, 0.f, 0.f, 0.f, 0.f, 0.f, 0.f, 0.f};
#pragma unroll
      for (int kc = 0; kc < 4; ++kc) {
        const int c = (q * 8 + wave) * 4 + kc;     // ntile = q*8+wave
        const uint16_t* bp = sWih + c * 512 + lane * 16;
        B32 tb;
        tb.lo = *(const uint4*)(bp);
        tb.hi = *(const uint4*)(bp + 8);
        const v16bf bm = __builtin_bit_cast(v16bf, tb);
        a = __builtin_amdgcn_wmma_f32_16x16x32_bf16(false, Ax[kc], false, bm,
                                                    (short)0, a, false, false);
      }
      acc[q] = a;
    }

    // ---- gates += h_{t-1} @ W_hh^T (16 WMMAs; B resident in VGPRs)
    const uint16_t* hr = sH + (t & 1) * BM * H + col * H;
    v16bf Ah[4];
#pragma unroll
    for (int kc = 0; kc < 4; ++kc) {
      B32 tmp;
      tmp.lo = *(const uint4*)(hr + kc * 32 + khalf * 8);
      tmp.hi = *(const uint4*)(hr + kc * 32 + 16 + khalf * 8);
      Ah[kc] = __builtin_bit_cast(v16bf, tmp);
    }
#pragma unroll
    for (int q = 0; q < 4; ++q)
#pragma unroll
      for (int kc = 0; kc < 4; ++kc)
        acc[q] = __builtin_amdgcn_wmma_f32_16x16x32_bf16(false, Ah[kc], false, Bhh[q][kc],
                                                         (short)0, acc[q], false, false);

    // ---- i,f,g,o share lane layout across the 4 accumulators -> pure VALU
    uint16_t* hw = sH + ((t + 1) & 1) * BM * H;
#pragma unroll
    for (int r = 0; r < 8; ++r) {
      const float iv = sigmf(acc[0][r] + bias[0]);
      const float fv = sigmf(acc[1][r] + bias[1]);
      const float gv = tanhf(acc[2][r] + bias[2]);
      const float ov = sigmf(acc[3][r] + bias[3]);
      const float cc = fv * cst[r] + iv * gv;
      cst[r] = cc;
      const float hv = ov * tanhf(cc);
      const uint16_t hb = f2bf_u16(hv);
      const int m = r + 8 * khalf;               // C/D layout: row of this vreg
      hw[m * H + n0 + col] = hb;                 // next step's A operand (LDS)
      hout[((size_t)t * BATCH + m0 + m) * H + n0 + col] = hb;  // layer output
    }
    __syncthreads();   // double-buffered sH: one barrier per timestep
  }
}

// ---------------------------------------------------------------------------
// Final FC head: out[b,t] = sigmoid(h3[t,b,:] . W_fc + b_fc)
// ---------------------------------------------------------------------------
__global__ __launch_bounds__(256)
void fc_head(const uint16_t* __restrict__ hin, const float* __restrict__ Wfc,
             const float* __restrict__ bfc, float* __restrict__ out)
{
  const int idx = blockIdx.x * blockDim.x + threadIdx.x;   // b*T + t
  if (idx >= BATCH * TSEQ) return;
  const int b = idx / TSEQ, t = idx % TSEQ;
  const uint16_t* hp = hin + ((size_t)t * BATCH + b) * H;
  float s = bfc[0];
#pragma unroll 8
  for (int k = 0; k < H; ++k) s += bf2f(hp[k]) * Wfc[k];
  out[idx] = 1.0f / (1.0f + __expf(-s));
}

extern "C" void kernel_launch(void* const* d_in, const int* in_sizes, int n_in,
                              void* d_out, int out_size, void* d_ws, size_t ws_size,
                              hipStream_t stream) {
  (void)in_sizes; (void)n_in; (void)out_size; (void)ws_size;
  const float* x    = (const float*)d_in[0];   // [B,T,IN0]
  const float* Wih0 = (const float*)d_in[1];   // [4H,IN0]
  const float* Wihr = (const float*)d_in[2];   // [3,4H,H]
  const float* Whh  = (const float*)d_in[3];   // [4,4H,H]
  const float* bih  = (const float*)d_in[4];   // [4,4H]
  const float* bhh  = (const float*)d_in[5];   // [4,4H]
  const float* Wfc  = (const float*)d_in[6];   // [1,H]
  const float* bfc  = (const float*)d_in[7];   // [1]
  float* out = (float*)d_out;

  const size_t NTB = (size_t)TSEQ * BATCH * H;   // elems per activation buffer
  uint16_t* xb = (uint16_t*)d_ws;                // bf16 padded x   (33.5 MB)
  uint16_t* h0 = xb + NTB;                       // bf16 ping       (33.5 MB)
  uint16_t* h1 = h0 + NTB;                       // bf16 pong       (33.5 MB)

  const size_t lds = 131072 + 2 * BM * H * sizeof(uint16_t);  // 136 KB / WGP
  dim3 grid(BATCH / BM), block(THREADS);

  x_to_bf16<<<(unsigned)((NTB + 255) / 256), 256, 0, stream>>>(x, xb);

  lstm_layer<<<grid, block, lds, stream>>>(xb, Wih0,              IN0,
      Whh + 0 * 512 * H, bih + 0,    bhh + 0,    h0);
  lstm_layer<<<grid, block, lds, stream>>>(h0, Wihr + 0 * 512 * H, H,
      Whh + 1 * 512 * H, bih + 512,  bhh + 512,  h1);
  lstm_layer<<<grid, block, lds, stream>>>(h1, Wihr + 1 * 512 * H, H,
      Whh + 2 * 512 * H, bih + 1024, bhh + 1024, h0);
  lstm_layer<<<grid, block, lds, stream>>>(h0, Wihr + 2 * 512 * H, H,
      Whh + 3 * 512 * H, bih + 1536, bhh + 1536, h1);

  fc_head<<<(BATCH * TSEQ + 255) / 256, 256, 0, stream>>>(h1, Wfc, bfc, out);
}
